// HyperedgeMessagePassingModule_73177652789992
// MI455X (gfx1250) — matrix operand order repaired and checked by reading, not compiled, not run on hardware
//
#include <hip/hip_runtime.h>

// ---------------------------------------------------------------------------
// Types
// ---------------------------------------------------------------------------
typedef __attribute__((ext_vector_type(16))) __bf16    v16bf;
typedef __attribute__((ext_vector_type(8)))  float     v8f;
typedef __attribute__((ext_vector_type(4)))  float     v4f;
typedef __attribute__((ext_vector_type(4)))  unsigned  v4u;
typedef __attribute__((ext_vector_type(2)))  unsigned  v2u;

union BF16x16 {
    v16bf v;
    v4u   q[2];
    unsigned short u16[16];
};

// ---------------------------------------------------------------------------
// Helpers
// ---------------------------------------------------------------------------
__device__ __forceinline__ unsigned short f2bf(float f) {
    unsigned u = __builtin_bit_cast(unsigned, f);
    u += 0x7fffu + ((u >> 16) & 1u);          // round-to-nearest-even
    return (unsigned short)(u >> 16);
}

__device__ __forceinline__ void store4bf(unsigned short* dst, v4f f) {
    union { unsigned short s[4]; v2u u; } p;
    p.s[0] = f2bf(f.x); p.s[1] = f2bf(f.y);
    p.s[2] = f2bf(f.z); p.s[3] = f2bf(f.w);
    *(v2u*)dst = p.u;                          // 8B-aligned ds_store_b64
}

__device__ __forceinline__ float leaky(float v) {
    return v > 0.0f ? v : 0.01f * v;
}

// Monotonic-bits float atomic max -> global_atomic_max_i32 / min_u32
__device__ __forceinline__ void atomicMaxF32(float* addr, float val) {
    if (val >= 0.0f)
        atomicMax((int*)addr, __builtin_bit_cast(int, val));
    else
        atomicMin((unsigned int*)addr, __builtin_bit_cast(unsigned int, val));
}

// ---------------------------------------------------------------------------
// Core per-wave WMMA GEMM: A tile in LDS (bf16, padded rows), B fragments
// pre-swizzled in global (bf16, L2-resident).
//   KT       : K / 32
//   ROWELEMS : padded LDS row length in bf16 elements
//   NTPW     : 16-col tiles per wave
// Wave covers rows [strip*16, strip*16+16) of the LDS tile and columns
// [cg*NTPW*16, ...).
// ---------------------------------------------------------------------------
template<int KT, int ROWELEMS, int NTPW>
__device__ __forceinline__ void wave_gemm(const unsigned short* __restrict__ ldsA,
                                          const unsigned short* __restrict__ wfrag,
                                          v8f* acc, int strip, int cg, int lane)
{
    const int m    = lane & 15;
    const int half = lane >> 4;
    const unsigned short* arow = ldsA + (strip * 16 + m) * ROWELEMS;
    for (int kt = 0; kt < KT; ++kt) {
        // A fragment: contiguous K order matching B fragments:
        //   lanes 0-15: K 0-15 ; lanes 16-31: K 16-31
        BF16x16 a;
        a.q[0] = *(const v4u*)(arow + kt * 32 + half * 16);      // ds_read_b128
        a.q[1] = *(const v4u*)(arow + kt * 32 + half * 16 + 8);  // ds_read_b128
        const unsigned short* bbase = wfrag + (size_t)(kt * 16 + cg * NTPW) * 512
                                            + lane * 16;
#pragma unroll
        for (int nt = 0; nt < NTPW; ++nt) {
            BF16x16 b;                                           // 2x global_load_b128
            b.q[0] = *(const v4u*)(bbase + nt * 512);
            b.q[1] = *(const v4u*)(bbase + nt * 512 + 8);
            acc[nt] = __builtin_amdgcn_wmma_f32_16x16x32_bf16(
                false, a.v, false, b.v, (short)0, acc[nt], false, false);
        }
    }
}

// ---------------------------------------------------------------------------
// Kernel 0: fp32 weights -> bf16 WMMA B-fragments in workspace.
// Fragment (kt,nt): element (lane,e) = W[kt*32 + (lane>>4)*16 + e][nt*16 + (lane&15)]
// stored at frag*512 + lane*16 + e  (32B contiguous per lane).
// ---------------------------------------------------------------------------
__global__ __launch_bounds__(256)
void swizzle_weights(const float* __restrict__ W1, const float* __restrict__ W2,
                     unsigned short* __restrict__ wsW1, unsigned short* __restrict__ wsW2)
{
    const int totalW1 = 192 * 512;             // 384x256
    const int total   = totalW1 + 320 * 512;   // + 640x256
    int tid = blockIdx.x * 256 + threadIdx.x;
    if (tid >= total) return;
    const float* W; unsigned short* dst; int id;
    if (tid < totalW1) { W = W1; dst = wsW1; id = tid; }
    else               { W = W2; dst = wsW2; id = tid - totalW1; }
    int frag = id >> 9, within = id & 511;
    int lane = within >> 4, e = within & 15;
    int kt = frag >> 4, nt = frag & 15;
    int k = kt * 32 + (lane >> 4) * 16 + e;
    int n = nt * 16 + (lane & 15);
    dst[id] = f2bf(W[k * 256 + n]);
}

// ---------------------------------------------------------------------------
// Kernel 1: edge-type rows.  msgs = leaky(type_reprs @ W1[0:256] + b1)
// Direct store into edge_states (seeds every segment of the segment-max).
// Tile: 64 rows x 256 cols.  8 waves: 4 row strips x 2 col groups (8 tiles/wave).
// ---------------------------------------------------------------------------
__global__ __launch_bounds__(256)
void gemm_edge(const float* __restrict__ type_reprs, const int* __restrict__ type_idx,
               const unsigned short* __restrict__ wsW1, const float* __restrict__ b1,
               float* __restrict__ edge_out, int E)
{
    __shared__ unsigned short sA[64 * 264];    // 33.8 KB, row pad -> bank-safe
    const int tid = threadIdx.x;
    const int blockRow = blockIdx.x * 64;

    for (int j = tid; j < 64 * 64; j += 256) { // 64 float4 per row
        int lr = j >> 6, v4 = j & 63;
        int gr = blockRow + lr; if (gr >= E) gr = E - 1;
        const v4f* src = (const v4f*)(type_reprs + (size_t)type_idx[gr] * 256);
        store4bf(&sA[lr * 264 + v4 * 4], src[v4]);
    }
    __syncthreads();

    const int w = tid >> 5, lane = tid & 31;
    const int strip = w >> 1, cg = w & 1;
    const int half = lane >> 4;
    v8f acc[8] = {};
    wave_gemm<8, 264, 8>(sA, wsW1, acc, strip, cg, lane);   // K = 256 only

#pragma unroll
    for (int nt = 0; nt < 8; ++nt) {
        int ncol = cg * 128 + nt * 16 + (lane & 15);
        float bi = b1[ncol];
#pragma unroll
        for (int i = 0; i < 8; ++i) {
            int gr = blockRow + strip * 16 + half * 8 + i;
            if (gr < E)
                edge_out[(size_t)gr * 256 + ncol] = leaky(acc[nt][i] + bi);
        }
    }
}

// ---------------------------------------------------------------------------
// Kernel 2: arg rows.  msgs = leaky([name|node] @ W1 + b1), K = 384,
// then fused segment-max: atomic float-max into edge_states[edge_id].
// Tile: 64 rows x 256 cols.
// ---------------------------------------------------------------------------
__global__ __launch_bounds__(256)
void gemm_arg1(const float* __restrict__ nodes, const int* __restrict__ node_idx,
               const float* __restrict__ names, const int* __restrict__ name_idx,
               const int* __restrict__ edge_id,
               const unsigned short* __restrict__ wsW1, const float* __restrict__ b1,
               float* __restrict__ edge_out, int A)
{
    __shared__ unsigned short sA[64 * 392];    // 50.2 KB
    __shared__ int s_eid[64];
    const int tid = threadIdx.x;
    const int blockRow = blockIdx.x * 64;

    if (tid < 64) {
        int gr = blockRow + tid; if (gr >= A) gr = A - 1;
        s_eid[tid] = edge_id[gr];
    }
    for (int j = tid; j < 64 * 96; j += 256) { // 96 float4 per row (128+256 cols)
        int lr = j / 96, v4 = j % 96;
        int gr = blockRow + lr; if (gr >= A) gr = A - 1;
        v4f f; int col;
        if (v4 < 32) {
            f = ((const v4f*)(names + (size_t)name_idx[gr] * 128))[v4];
            col = v4 * 4;
        } else {
            f = ((const v4f*)(nodes + (size_t)node_idx[gr] * 256))[v4 - 32];
            col = 128 + (v4 - 32) * 4;
        }
        store4bf(&sA[lr * 392 + col], f);
    }
    __syncthreads();

    const int w = tid >> 5, lane = tid & 31;
    const int strip = w >> 1, cg = w & 1;
    const int half = lane >> 4;
    v8f acc[8] = {};
    wave_gemm<12, 392, 8>(sA, wsW1, acc, strip, cg, lane);  // K = 384

#pragma unroll
    for (int nt = 0; nt < 8; ++nt) {
        int ncol = cg * 128 + nt * 16 + (lane & 15);
        float bi = b1[ncol];
#pragma unroll
        for (int i = 0; i < 8; ++i) {
            int lr = strip * 16 + half * 8 + i;
            int gr = blockRow + lr;
            if (gr < A) {
                float v = leaky(acc[nt][i] + bi);
                atomicMaxF32(&edge_out[(size_t)s_eid[lr] * 256 + ncol], v);
            }
        }
    }
}

// ---------------------------------------------------------------------------
// Kernel 3: update MLP. out = leaky([edge_states[eid]|name|node] @ W2 + b2),
// K = 640.  Tile: 32 rows x 256 cols. 8 waves: 2 strips x 4 col groups
// (4 tiles/wave).
// ---------------------------------------------------------------------------
__global__ __launch_bounds__(256)
void gemm_arg2(const float* __restrict__ nodes, const int* __restrict__ node_idx,
               const float* __restrict__ names, const int* __restrict__ name_idx,
               const int* __restrict__ edge_id, const float* __restrict__ edge_states,
               const unsigned short* __restrict__ wsW2, const float* __restrict__ b2,
               float* __restrict__ out_msg, int A)
{
    __shared__ unsigned short sA[32 * 648];    // 41.5 KB
    const int tid = threadIdx.x;
    const int blockRow = blockIdx.x * 32;

    for (int j = tid; j < 32 * 160; j += 256) { // 160 float4 per row (256+128+256)
        int lr = j / 160, v4 = j % 160;
        int gr = blockRow + lr; if (gr >= A) gr = A - 1;
        v4f f; int col;
        if (v4 < 64) {
            f = ((const v4f*)(edge_states + (size_t)edge_id[gr] * 256))[v4];
            col = v4 * 4;
        } else if (v4 < 96) {
            f = ((const v4f*)(names + (size_t)name_idx[gr] * 128))[v4 - 64];
            col = 256 + (v4 - 64) * 4;
        } else {
            f = ((const v4f*)(nodes + (size_t)node_idx[gr] * 256))[v4 - 96];
            col = 384 + (v4 - 96) * 4;
        }
        store4bf(&sA[lr * 648 + col], f);
    }
    __syncthreads();

    const int w = tid >> 5, lane = tid & 31;
    const int strip = w >> 2, cg = w & 3;
    const int half = lane >> 4;
    v8f acc[4] = {};
    wave_gemm<20, 648, 4>(sA, wsW2, acc, strip, cg, lane);  // K = 640

#pragma unroll
    for (int nt = 0; nt < 4; ++nt) {
        int ncol = cg * 64 + nt * 16 + (lane & 15);
        float bi = b2[ncol];
#pragma unroll
        for (int i = 0; i < 8; ++i) {
            int gr = blockRow + strip * 16 + half * 8 + i;
            if (gr < A)
                out_msg[(size_t)gr * 256 + ncol] = leaky(acc[nt][i] + bi);
        }
    }
}

// ---------------------------------------------------------------------------
// Launch
// ---------------------------------------------------------------------------
extern "C" void kernel_launch(void* const* d_in, const int* in_sizes, int n_in,
                              void* d_out, int out_size, void* d_ws, size_t ws_size,
                              hipStream_t stream) {
    const float* nodes      = (const float*)d_in[0];
    const int*   node_idx   = (const int*)  d_in[1];
    const float* type_reprs = (const float*)d_in[2];
    const int*   type_idx   = (const int*)  d_in[3];
    const float* names      = (const float*)d_in[4];
    const int*   name_idx   = (const int*)  d_in[5];
    const int*   edge_id    = (const int*)  d_in[6];
    const float* W1         = (const float*)d_in[8];
    const float* b1         = (const float*)d_in[9];
    const float* W2         = (const float*)d_in[10];
    const float* b2         = (const float*)d_in[11];

    const int A = in_sizes[1];   // 250000 args
    const int E = in_sizes[3];   // 60000 edges

    float* out_msg  = (float*)d_out;                 // [A,256]
    float* edge_out = out_msg + (size_t)A * 256;     // [E,256]

    unsigned short* wsW1 = (unsigned short*)d_ws;    // 192 frags * 1KB
    unsigned short* wsW2 = wsW1 + 192 * 512;         // 320 frags * 1KB

    // 1) weights -> bf16 WMMA fragments (L2 resident afterwards)
    swizzle_weights<<<1024, 256, 0, stream>>>(W1, W2, wsW1, wsW2);
    // 2) seed edge_states with edge-type messages (direct store)
    gemm_edge<<<(E + 63) / 64, 256, 0, stream>>>(type_reprs, type_idx, wsW1, b1,
                                                 edge_out, E);
    // 3) arg messages + fused atomic segment-max into edge_states
    gemm_arg1<<<(A + 63) / 64, 256, 0, stream>>>(nodes, node_idx, names, name_idx,
                                                 edge_id, wsW1, b1, edge_out, A);
    // 4) update MLP -> message_to_nodes
    gemm_arg2<<<(A + 31) / 32, 256, 0, stream>>>(nodes, node_idx, names, name_idx,
                                                 edge_id, edge_out, wsW2, b2,
                                                 out_msg, A);
}
